// SHGNN_62526133895565
// MI455X (gfx1250) — matrix operand order repaired and checked by reading, not compiled, not run on hardware
//
#include <hip/hip_runtime.h>
#include <hip/hip_bf16.h>

// ---------------------------------------------------------------------------
// SHGNN (AllSet hypergraph transformer) forward for gfx1250 / MI455X.
// GEMMs run through v_wmma_f32_16x16x32_bf16 with vectorized fragment loads:
//   A: LDS-staged 32x256 bf16 tile, 2x ds_load_b128 per fragment (natural
//      row-major layout IS fragment order: two contiguous 8-elem K-runs).
//   B: weights pre-transposed to bf16 [n][k]; fragment = two 8-elem
//      K-runs (k0+8g, k0+16+8g), 2x 16B global loads per lane, reused across 2 row-tiles (L2/WGP$ resident).
// Pooling is a deterministic segment-softmax (sorted targets -> CSR offsets).
// ---------------------------------------------------------------------------

typedef __attribute__((ext_vector_type(16))) __bf16 v16bf;
typedef __attribute__((ext_vector_type(8)))  __bf16 v8bf;
typedef __attribute__((ext_vector_type(8)))  float  v8f;

#define D_DIM 256
#define HEADS 4
#define DH 64
#define ROWS_PB 32           // rows per GEMM block (2 row-tiles of 16)

static __device__ inline unsigned short f32_to_bf16(float f) {
    unsigned u = __float_as_uint(f);
    unsigned r = u + 0x7FFFu + ((u >> 16) & 1u);   // round-to-nearest-even
    return (unsigned short)(r >> 16);
}

// ---------------------------------------------------------------------------
// fp32 [k][n] -> bf16 transposed [n][k] weight conversion (once per launch)
// ---------------------------------------------------------------------------
__global__ void f2bf_transpose_kernel(const float* __restrict__ src,
                                      unsigned short* __restrict__ dst,
                                      int nmats) {
    int total = nmats * D_DIM * D_DIM;
    for (int i = blockIdx.x * blockDim.x + threadIdx.x; i < total;
         i += gridDim.x * blockDim.x) {
        int m = i >> 16;
        int r = i & 65535;
        int n = r & 255, k = r >> 8;          // consecutive i -> consecutive n (coalesced read)
        dst[(size_t)m * 65536 + (size_t)n * D_DIM + k] =
            f32_to_bf16(src[(size_t)m * 65536 + (size_t)k * D_DIM + n]);
    }
}

// ---------------------------------------------------------------------------
// BatchNorm (training stats over N rows, per feature)
// ---------------------------------------------------------------------------
__global__ void bn_stats_kernel(const float* __restrict__ x,
                                float* __restrict__ meanvar, int Nn) {
    int d = blockIdx.x;          // feature 0..255
    int t = threadIdx.x;         // 0..255
    float s1 = 0.f, s2 = 0.f;
    for (int n = t; n < Nn; n += 256) {
        float v = x[(size_t)n * D_DIM + d];
        s1 += v; s2 += v * v;
    }
    __shared__ float r1[256], r2[256];
    r1[t] = s1; r2[t] = s2; __syncthreads();
    for (int off = 128; off >= 1; off >>= 1) {
        if (t < off) { r1[t] += r1[t + off]; r2[t] += r2[t + off]; }
        __syncthreads();
    }
    if (t == 0) {
        float m = r1[0] / (float)Nn;
        meanvar[d]         = m;
        meanvar[D_DIM + d] = r2[0] / (float)Nn - m * m;
    }
}

__global__ void bn_norm_kernel(const float* __restrict__ x,
                               const float* __restrict__ meanvar,
                               const float* __restrict__ g,
                               const float* __restrict__ b,
                               float* __restrict__ y, int total) {
    int i = blockIdx.x * blockDim.x + threadIdx.x;
    if (i >= total) return;
    int d = i & (D_DIM - 1);
    float m = meanvar[d], v = meanvar[D_DIM + d];
    y[i] = (x[i] - m) * rsqrtf(v + 1e-5f) * g[d] + b[d];
}

// ---------------------------------------------------------------------------
// Segment CSR offsets from a sorted target array (deterministic, no atomics)
// ---------------------------------------------------------------------------
__global__ void seg_starts_kernel(const int* __restrict__ tgt,
                                  int* __restrict__ starts, int M, int nseg) {
    int m = blockIdx.x * blockDim.x + threadIdx.x;
    if (m >= M) return;
    int cur = tgt[m];
    if (m == 0) {
        for (int s = 0; s <= cur; ++s) starts[s] = 0;
    } else {
        int prev = tgt[m - 1];
        for (int s = prev + 1; s <= cur; ++s) starts[s] = m;
    }
    if (m == M - 1)
        for (int s = cur + 1; s <= nseg; ++s) starts[s] = M;
}

// ---------------------------------------------------------------------------
// alpha[r,h] = dot(Kproj[r, h*64 : h*64+64], seed[h]) / sqrt(64)
// ---------------------------------------------------------------------------
__global__ void alpha_kernel(const float* __restrict__ Kp,
                             const float* __restrict__ seed,
                             float* __restrict__ alphaN, int R) {
    int i = blockIdx.x * blockDim.x + threadIdx.x;
    if (i >= R * HEADS) return;
    int r = i >> 2, h = i & 3;
    const float* kr = Kp + (size_t)r * D_DIM + h * DH;
    const float* sr = seed + h * DH;
    float s = 0.f;
    #pragma unroll 8
    for (int d = 0; d < DH; ++d) s += kr[d] * sr[d];
    alphaN[i] = s * 0.125f;   // 1/sqrt(64)
}

// ---------------------------------------------------------------------------
// Segment softmax-weighted pooling: out[s,:] = sum_m w(m,h) * V[src[m],:] + seed
// ---------------------------------------------------------------------------
__global__ void seg_pool_kernel(const int* __restrict__ starts,
                                const int* __restrict__ src,
                                const float* __restrict__ alphaN,  // [R,4]
                                const float* __restrict__ V,       // [R,256]
                                const float* __restrict__ seed,    // [256]
                                float* __restrict__ out, int nseg) {
    int s = blockIdx.x;
    int t = threadIdx.x;            // 0..255 -> channel
    int h = t >> 6;                 // head for my channel
    int b = starts[s], e = starts[s + 1];

    if (b >= e) { out[(size_t)s * D_DIM + t] = seed[t]; return; }

    __shared__ float red[256];
    __shared__ float amax[HEADS], den[HEADS];
    __shared__ int   sidx[64];
    __shared__ float wgt[64 * HEADS];

    int hh = t & 3;                 // head for reduction role
    int j  = t >> 2;                // 0..63
    // per-head max
    float pm = -3.0e38f;
    for (int m = b + j; m < e; m += 64)
        pm = fmaxf(pm, alphaN[(size_t)src[m] * HEADS + hh]);
    red[t] = pm; __syncthreads();
    for (int off = 128; off >= 4; off >>= 1) {
        if (t < off) red[t] = fmaxf(red[t], red[t + off]);
        __syncthreads();
    }
    if (t < HEADS) amax[t] = red[t];
    __syncthreads();
    // per-head denominator
    float ps = 0.f;
    for (int m = b + j; m < e; m += 64)
        ps += __expf(alphaN[(size_t)src[m] * HEADS + hh] - amax[hh]);
    red[t] = ps; __syncthreads();
    for (int off = 128; off >= 4; off >>= 1) {
        if (t < off) red[t] += red[t + off];
        __syncthreads();
    }
    if (t < HEADS) den[t] = fmaxf(red[t], 1e-9f);
    __syncthreads();

    float am = amax[h], dn = den[h];
    float acc = 0.f;
    for (int mb = b; mb < e; mb += 64) {
        int cnt = min(64, e - mb);
        if (t < cnt) sidx[t] = src[mb + t];
        if (j < cnt)
            wgt[j * HEADS + hh] =
                __expf(alphaN[(size_t)src[mb + j] * HEADS + hh] - am) / dn;
        __syncthreads();
        for (int q = 0; q < cnt; ++q)
            acc += wgt[q * HEADS + h] * V[(size_t)sidx[q] * D_DIM + t];
        __syncthreads();
    }
    out[(size_t)s * D_DIM + t] = acc + seed[t];
}

// ---------------------------------------------------------------------------
// LayerNorm over 256 channels (one block per row)
// ---------------------------------------------------------------------------
__global__ void ln_kernel(const float* __restrict__ in, float* __restrict__ out,
                          const float* __restrict__ g, const float* __restrict__ b) {
    __shared__ float red[256];
    __shared__ float sm, sv;
    int r = blockIdx.x, t = threadIdx.x;
    float v = in[(size_t)r * D_DIM + t];
    red[t] = v; __syncthreads();
    for (int off = 128; off >= 1; off >>= 1) {
        if (t < off) red[t] += red[t + off];
        __syncthreads();
    }
    if (t == 0) sm = red[0] * (1.f / 256.f);
    __syncthreads();
    float d0 = v - sm;
    red[t] = d0 * d0; __syncthreads();
    for (int off = 128; off >= 1; off >>= 1) {
        if (t < off) red[t] += red[t + off];
        __syncthreads();
    }
    if (t == 0) sv = red[0] * (1.f / 256.f);
    __syncthreads();
    out[(size_t)r * D_DIM + t] = d0 * rsqrtf(sv + 1e-5f) * g[t] + b[t];
}

// out = relu( LN(a + c) )
__global__ void addln_relu_kernel(const float* __restrict__ a,
                                  const float* __restrict__ c,
                                  float* __restrict__ out,
                                  const float* __restrict__ g,
                                  const float* __restrict__ b) {
    __shared__ float red[256];
    __shared__ float sm, sv;
    int r = blockIdx.x, t = threadIdx.x;
    float v = a[(size_t)r * D_DIM + t] + c[(size_t)r * D_DIM + t];
    red[t] = v; __syncthreads();
    for (int off = 128; off >= 1; off >>= 1) {
        if (t < off) red[t] += red[t + off];
        __syncthreads();
    }
    if (t == 0) sm = red[0] * (1.f / 256.f);
    __syncthreads();
    float d0 = v - sm;
    red[t] = d0 * d0; __syncthreads();
    for (int off = 128; off >= 1; off >>= 1) {
        if (t < off) red[t] += red[t + off];
        __syncthreads();
    }
    if (t == 0) sv = red[0] * (1.f / 256.f);
    __syncthreads();
    float o = d0 * rsqrtf(sv + 1e-5f) * g[t] + b[t];
    out[(size_t)r * D_DIM + t] = fmaxf(o, 0.f);
}

// ---------------------------------------------------------------------------
// WMMA GEMM: Y[R x 256] = act( X[R x 256] @ W[256 x 256] + bias )
// W supplied pre-transposed as bf16 Wt[n][k].
// block = 128 threads (4 waves); block tile = 32 rows x 256 cols;
// each wave: 2 row-tiles x 4 col-tiles -> each B fragment feeds 2 WMMAs.
// ---------------------------------------------------------------------------
__global__ __launch_bounds__(128)
void gemm256_wmma_kernel(const float* __restrict__ X,
                         const unsigned short* __restrict__ Wt,  // bf16 bits [n][k]
                         const float* __restrict__ bias,
                         float* __restrict__ Y, int R, int relu) {
    __shared__ unsigned short As[ROWS_PB][264];  // 32x256 bf16 A tile (+pad, 16B-aligned rows)

    int tid  = threadIdx.x;
    int wave = tid >> 5;
    int lane = tid & 31;
    int r0   = blockIdx.x * ROWS_PB;
    int col0 = wave * 64;
    int m16  = lane & 15;
    int g    = lane >> 4;

    // prefetch next block's A rows (emits global_prefetch_b8)
    if ((size_t)(r0 + ROWS_PB) < (size_t)R)
        __builtin_prefetch(X + (size_t)(r0 + ROWS_PB) * D_DIM + tid * 16, 0, 0);

    // stage A tile once: pack 2 bf16 per 32-bit LDS store, coalesced X reads
    for (int i = tid; i < ROWS_PB * 128; i += 128) {
        int rr = i >> 7;              // row 0..31
        int kk = (i & 127) * 2;       // col pair
        int row = r0 + rr;
        float v0 = 0.f, v1 = 0.f;
        if (row < R) {
            v0 = X[(size_t)row * D_DIM + kk];
            v1 = X[(size_t)row * D_DIM + kk + 1];
        }
        unsigned pack = (unsigned)f32_to_bf16(v0) | ((unsigned)f32_to_bf16(v1) << 16);
        *(unsigned*)&As[rr][kk] = pack;
    }
    __syncthreads();

    v8f acc[2][4] = {};

    #pragma unroll
    for (int k0 = 0; k0 < D_DIM; k0 += 32) {
        // A fragments (both row-tiles): two contiguous 8-elem K-runs each
        v16bf afrag[2];
        #pragma unroll
        for (int rt = 0; rt < 2; ++rt) {
            v8bf alo = *(const v8bf*)&As[rt * 16 + m16][k0 + g * 8];
            v8bf ahi = *(const v8bf*)&As[rt * 16 + m16][k0 + 16 + g * 8];
            afrag[rt] = __builtin_shufflevector(alo, ahi,
                0, 1, 2, 3, 4, 5, 6, 7, 8, 9, 10, 11, 12, 13, 14, 15);
        }
        #pragma unroll
        for (int st = 0; st < 4; ++st) {
            int n = col0 + st * 16 + m16;
            // B fragment: two 8-elem K-runs (k0+8g, k0+16+8g) in transposed weights,
            // reused by both row-tiles
            v8bf blo = *(const v8bf*)(Wt + (size_t)n * D_DIM + k0 + g * 8);
            v8bf bhi = *(const v8bf*)(Wt + (size_t)n * D_DIM + k0 + 16 + g * 8);
            v16bf bfrag = __builtin_shufflevector(blo, bhi,
                0, 1, 2, 3, 4, 5, 6, 7, 8, 9, 10, 11, 12, 13, 14, 15);
            #pragma unroll
            for (int rt = 0; rt < 2; ++rt)
                acc[rt][st] = __builtin_amdgcn_wmma_f32_16x16x32_bf16(
                    false, afrag[rt], false, bfrag, (short)0, acc[rt][st],
                    false, false);
        }
    }

    // C/D layout: VGPR rv -> row rv + 8*g ; col = lane%16
    #pragma unroll
    for (int rt = 0; rt < 2; ++rt) {
        #pragma unroll
        for (int st = 0; st < 4; ++st) {
            #pragma unroll
            for (int rv = 0; rv < 8; ++rv) {
                int row = r0 + rt * 16 + rv + 8 * g;
                int col = col0 + st * 16 + m16;
                if (row < R) {
                    float v = acc[rt][st][rv] + bias[col];
                    if (relu) v = fmaxf(v, 0.f);
                    Y[(size_t)row * D_DIM + col] = v;
                }
            }
        }
    }
}

// ---------------------------------------------------------------------------
// Classifier: logits = [xL0|xL1] @ Wc[512,10] + bc ; log_softmax.
// ---------------------------------------------------------------------------
__global__ void classifier_kernel(const float* __restrict__ xL0,
                                  const float* __restrict__ xL1,
                                  const float* __restrict__ Wc,
                                  const float* __restrict__ bc,
                                  float* __restrict__ out, int Nn) {
    int gw   = (blockIdx.x * blockDim.x + threadIdx.x) >> 5;
    int lane = threadIdx.x & 31;
    if (gw >= Nn) return;
    float acc[10];
    #pragma unroll
    for (int c = 0; c < 10; ++c) acc[c] = 0.f;
    for (int k = lane; k < 2 * D_DIM; k += 32) {
        float v = (k < D_DIM) ? xL0[(size_t)gw * D_DIM + k]
                              : xL1[(size_t)gw * D_DIM + (k - D_DIM)];
        const float* wr = Wc + (size_t)k * 10;
        #pragma unroll
        for (int c = 0; c < 10; ++c) acc[c] += v * wr[c];
    }
    #pragma unroll
    for (int off = 16; off > 0; off >>= 1)
        #pragma unroll
        for (int c = 0; c < 10; ++c) acc[c] += __shfl_down(acc[c], off, 32);
    if (lane == 0) {
        float mx = -3.0e38f;
        #pragma unroll
        for (int c = 0; c < 10; ++c) { acc[c] += bc[c]; mx = fmaxf(mx, acc[c]); }
        float lse = 0.f;
        #pragma unroll
        for (int c = 0; c < 10; ++c) lse += __expf(acc[c] - mx);
        lse = __logf(lse) + mx;
        #pragma unroll
        for (int c = 0; c < 10; ++c) out[(size_t)gw * 10 + c] = acc[c] - lse;
    }
}

// ---------------------------------------------------------------------------
// Host orchestration
// ---------------------------------------------------------------------------
extern "C" void kernel_launch(void* const* d_in, const int* in_sizes, int n_in,
                              void* d_out, int out_size, void* d_ws, size_t ws_size,
                              hipStream_t stream) {
    (void)n_in; (void)out_size; (void)ws_size;

    const float* node_x    = (const float*)d_in[0];
    const int*   nodes_map = (const int*)  d_in[1];
    const int*   e_batch   = (const int*)  d_in[2];
    const int*   edges_map = (const int*)  d_in[3];
    const int*   n_batch   = (const int*)  d_in[4];
    const float* bn_g      = (const float*)d_in[5];
    const float* bn_b      = (const float*)d_in[6];
    const float* Wc        = (const float*)d_in[7];
    const float* bc        = (const float*)d_in[8];
    const float* Wk_ne = (const float*)d_in[9];  const float* bk_ne = (const float*)d_in[10];
    const float* Wv_ne = (const float*)d_in[11]; const float* bv_ne = (const float*)d_in[12];
    const float* sd_ne = (const float*)d_in[13];
    const float* Wff_ne = (const float*)d_in[14]; const float* bff_ne = (const float*)d_in[15];
    const float* g0_ne = (const float*)d_in[16]; const float* b0_ne = (const float*)d_in[17];
    const float* g1_ne = (const float*)d_in[18]; const float* b1_ne = (const float*)d_in[19];
    const float* Wk_en = (const float*)d_in[20]; const float* bk_en = (const float*)d_in[21];
    const float* Wv_en = (const float*)d_in[22]; const float* bv_en = (const float*)d_in[23];
    const float* sd_en = (const float*)d_in[24];
    const float* Wff_en = (const float*)d_in[25]; const float* bff_en = (const float*)d_in[26];
    const float* g0_en = (const float*)d_in[27]; const float* b0_en = (const float*)d_in[28];
    const float* g1_en = (const float*)d_in[29]; const float* b1_en = (const float*)d_in[30];

    const int Nn = in_sizes[0] / D_DIM;   // 100000
    const int M  = in_sizes[1];           // 500000
    const int En = 25000;                 // reference constant
    const int MAT = D_DIM * D_DIM;        // 65536

    // ---- workspace bump allocator ----
    char* p = (char*)d_ws;
    auto alloc = [&](size_t bytes) -> void* {
        void* r = (void*)p;
        p += (bytes + 255) & ~((size_t)255);
        return r;
    };
    float* x0     = (float*)alloc((size_t)Nn * D_DIM * 4);
    float* xL0    = (float*)alloc((size_t)Nn * D_DIM * 4);
    float* xL1    = (float*)alloc((size_t)Nn * D_DIM * 4);
    float* projK  = (float*)alloc((size_t)Nn * D_DIM * 4);
    float* projV  = (float*)alloc((size_t)Nn * D_DIM * 4);
    float* t1     = (float*)alloc((size_t)Nn * D_DIM * 4);
    float* t2     = (float*)alloc((size_t)Nn * D_DIM * 4);
    float* edgex  = (float*)alloc((size_t)En * D_DIM * 4);
    float* alphaN = (float*)alloc((size_t)Nn * HEADS * 4);
    float* meanvar= (float*)alloc(2 * D_DIM * 4);
    int*   st_e   = (int*)alloc((size_t)(En + 1) * 4);
    int*   st_n   = (int*)alloc((size_t)(Nn + 1) * 4);
    unsigned short* wbf = (unsigned short*)alloc((size_t)6 * 2 * MAT * 2);
    // wbf layout: [mat 0..5][layer 0..1][65536] (bf16, transposed [n][k])

    // ---- one-time weight conversion: fp32 [k][n] -> bf16 transposed [n][k] ----
    const float* wsrc[6] = {Wk_ne, Wv_ne, Wff_ne, Wk_en, Wv_en, Wff_en};
    for (int i = 0; i < 6; ++i)
        f2bf_transpose_kernel<<<512, 256, 0, stream>>>(
            wsrc[i], wbf + (size_t)i * 2 * MAT, 2);

    // ---- input BatchNorm ----
    bn_stats_kernel<<<D_DIM, 256, 0, stream>>>(node_x, meanvar, Nn);
    bn_norm_kernel<<<(Nn * D_DIM + 255) / 256, 256, 0, stream>>>(
        node_x, meanvar, bn_g, bn_b, x0, Nn * D_DIM);

    // ---- segment offsets (sorted targets) ----
    seg_starts_kernel<<<(M + 255) / 256, 256, 0, stream>>>(e_batch, st_e, M, En);
    seg_starts_kernel<<<(M + 255) / 256, 256, 0, stream>>>(n_batch, st_n, M, Nn);

    // ---- PMA stage helper ----
    auto run_pma = [&](const float* xin, int Rin,
                       const int* starts, const int* src, int nseg,
                       const unsigned short* WkB, const float* bk,
                       const unsigned short* WvB, const float* bv,
                       const float* seed,
                       const unsigned short* WffB, const float* bff,
                       const float* g0, const float* b0,
                       const float* g1, const float* b1,
                       float* dest) {
        int gR = (Rin + ROWS_PB - 1) / ROWS_PB;
        gemm256_wmma_kernel<<<gR, 128, 0, stream>>>(xin, WkB, bk, projK, Rin, 0);
        gemm256_wmma_kernel<<<gR, 128, 0, stream>>>(xin, WvB, bv, projV, Rin, 0);
        alpha_kernel<<<(Rin * HEADS + 255) / 256, 256, 0, stream>>>(projK, seed, alphaN, Rin);
        seg_pool_kernel<<<nseg, 256, 0, stream>>>(starts, src, alphaN, projV, seed, t1, nseg);
        ln_kernel<<<nseg, 256, 0, stream>>>(t1, t2, g0, b0);
        gemm256_wmma_kernel<<<(nseg + ROWS_PB - 1) / ROWS_PB, 128, 0, stream>>>(
            t2, WffB, bff, t1, nseg, 1);
        addln_relu_kernel<<<nseg, 256, 0, stream>>>(t2, t1, dest, g1, b1);
    };

    const float* cur = x0;
    for (int l = 0; l < 2; ++l) {
        // N2E: pool nodes into hyperedges
        run_pma(cur, Nn, st_e, nodes_map, En,
                wbf + (size_t)(0 * 2 + l) * MAT, bk_ne + l * D_DIM,
                wbf + (size_t)(1 * 2 + l) * MAT, bv_ne + l * D_DIM,
                sd_ne + l * D_DIM,
                wbf + (size_t)(2 * 2 + l) * MAT, bff_ne + l * D_DIM,
                g0_ne + l * D_DIM, b0_ne + l * D_DIM,
                g1_ne + l * D_DIM, b1_ne + l * D_DIM,
                edgex);
        // E2N: pool hyperedges into nodes
        float* dst = (l == 0) ? xL0 : xL1;
        run_pma(edgex, En, st_n, edges_map, Nn,
                wbf + (size_t)(3 * 2 + l) * MAT, bk_en + l * D_DIM,
                wbf + (size_t)(4 * 2 + l) * MAT, bv_en + l * D_DIM,
                sd_en + l * D_DIM,
                wbf + (size_t)(5 * 2 + l) * MAT, bff_en + l * D_DIM,
                g0_en + l * D_DIM, b0_en + l * D_DIM,
                g1_en + l * D_DIM, b1_en + l * D_DIM,
                dst);
        cur = dst;
    }

    // ---- classifier + log_softmax ----
    classifier_kernel<<<(Nn + 7) / 8, 256, 0, stream>>>(
        xL0, xL1, Wc, bc, (float*)d_out, Nn);
}
